// Block_55413668053643
// MI455X (gfx1250) — matrix operand order repaired and checked by reading, not compiled, not run on hardware
//
#include <hip/hip_runtime.h>

#define DEVINL __device__ __forceinline__

typedef __bf16 bf16_t;
typedef __bf16 v16bf __attribute__((ext_vector_type(16)));
typedef float  v8f   __attribute__((ext_vector_type(8)));
typedef unsigned int u32x4 __attribute__((ext_vector_type(4)));
typedef int i32x4 __attribute__((ext_vector_type(4)));
typedef int i32x8 __attribute__((ext_vector_type(8)));

static constexpr int Bsz = 2, Tn = 2048, Cn = 1024, Hn = 16, DHn = 64, HIDn = 4096;
static constexpr int BT = Bsz * Tn;

// ---------------- helpers ----------------

DEVINL bf16_t f2bf(float f) { return (bf16_t)f; }  // native v_cvt_pk_bf16_f32

DEVINL v8f vzero8() {
  v8f z;
#pragma unroll
  for (int e = 0; e < 8; ++e) z[e] = 0.0f;
  return z;
}

// A-matrix 16x32 bf16 fragment, row-major source (ISA 7.12.2):
// lane&15 = row M; lane>>4 selects K half {0..7,16..23} vs {8..15,24..31}
// Per lane: two contiguous 16B runs -> 2x ds_load_b128.
DEVINL v16bf load_frag_a(const bf16_t* __restrict__ src, int lda) {
  const int lane = threadIdx.x & 31;
  const int row = lane & 15, hf = lane >> 4;
  v16bf a;
#pragma unroll
  for (int i = 0; i < 8; ++i) {
    const int kk = ((i < 4) ? (2 * i) : (8 + 2 * i)) + 8 * hf;
    a[2 * i]     = src[row * lda + kk];
    a[2 * i + 1] = src[row * lda + kk + 1];
  }
  return a;
}

// B-matrix 32x16 fragment from a TRANSPOSED (n-major) source:
// logical element (k,n) = src[n*ldt + k]. Per lane: K {0..7,16..23} vs {8..15,24..31} -> 2x b128.
DEVINL v16bf load_frag_bt(const bf16_t* __restrict__ src, int ldt) {
  const int lane = threadIdx.x & 31;
  const int col = lane & 15, hf = lane >> 4;
  v16bf b;
#pragma unroll
  for (int i = 0; i < 8; ++i) {
    const int k0 = ((i < 4) ? (2 * i) : (8 + 2 * i)) + 8 * hf;
    b[2 * i]     = src[col * ldt + k0];
    b[2 * i + 1] = src[(col * ldt + k0) + 1];
  }
  return b;
}

DEVINL v8f wmma_bf16(v16bf a, v16bf b, v8f c) {
  return __builtin_amdgcn_wmma_f32_16x16x32_bf16(false, a, false, b, (short)0, c,
                                                 false, false);
}

union U4B16 {
  uint4 u;
  bf16_t e[8];
};

// ---------------- Tensor Data Mover: 2D tile -> LDS (ISA ch.8 D#) ----------------
// Copies tile_rows x tile_k (bf16) from a row-major tensor (row stride
// stride_elems) into LDS at lds_off, inserting 16B of padding after every 64B
// row (so LDS row stride = tile_k + 8 = 40 elements). Tracked by TENSORcnt.

DEVINL void tdm_load_tile(unsigned lds_off, const bf16_t* gaddr, int tensor_k,
                          int tensor_rows, int tile_k, int tile_rows,
                          int stride_elems) {
  const unsigned long long ga = (unsigned long long)(uintptr_t)gaddr;
  u32x4 g0;
  g0[0] = 1u;                                              // count=1 (valid), user mode
  g0[1] = lds_off;                                         // lds_addr (bytes)
  g0[2] = (unsigned)(ga & 0xFFFFFFFFu);                    // global_addr[31:0]
  g0[3] = (unsigned)((ga >> 32) & 0x1FFFFFFu) | (2u << 30);  // addr[56:32] | type=2
  i32x8 g1;
  // data_size=1 (2B) | pad_enable | pad_interval=3 (16 DW=64B) | pad_amount=3 (4 DW=16B)
  g1[0] = (1 << 16) | (1 << 20) | (3 << 22) | (3 << 25);
  g1[1] = (tensor_k & 0xFFFF) << 16;                       // tensor_dim0[15:0]
  g1[2] = (int)(((unsigned)tensor_k >> 16) | ((unsigned)(tensor_rows & 0xFFFF) << 16));
  g1[3] = (int)(((unsigned)tensor_rows >> 16) | ((unsigned)tile_k << 16));  // tile_dim0
  g1[4] = tile_rows;                                       // tile_dim1 (tile_dim2=0)
  g1[5] = stride_elems;                                    // tensor_dim0_stride[31:0]
  g1[6] = 0;
  g1[7] = 0;
  i32x4 z4 = {0, 0, 0, 0};
#if __clang_major__ >= 23
  i32x8 z8 = {0, 0, 0, 0, 0, 0, 0, 0};
  __builtin_amdgcn_tensor_load_to_lds(g0, g1, z4, z4, z8, 0);
#else
  __builtin_amdgcn_tensor_load_to_lds(g0, g1, z4, z4, 0);
#endif
}

DEVINL void tdm_wait0() { __builtin_amdgcn_s_wait_tensorcnt((short)0); }

// ---------------- weight conversion / transpose kernels ----------------

__global__ void cvt_bf16_kernel(const float* __restrict__ in, bf16_t* __restrict__ out,
                                int n) {
  int i = blockIdx.x * blockDim.x + threadIdx.x;
  const int stride = gridDim.x * blockDim.x;
  for (; i < n; i += stride) out[i] = f2bf(in[i]);
}

// in: [K][N] f32 row-major  ->  out: [N][K] bf16 row-major (n-major for GEMM B)
__global__ void tr2d_bf16_kernel(const float* __restrict__ in, bf16_t* __restrict__ out,
                                 int K, int N) {
  int i = blockIdx.x * blockDim.x + threadIdx.x;
  const int stride = gridDim.x * blockDim.x;
  const int total = K * N;
  for (; i < total; i += stride) {
    const int n = i / K, kk = i - n * K;
    out[i] = f2bf(in[(size_t)kk * N + n]);
  }
}

// in: [H][C][DH] f32  ->  out: [H][DH][C] bf16 (per-head n-major)
__global__ void tr_head_bf16_kernel(const float* __restrict__ in,
                                    bf16_t* __restrict__ out) {
  int i = blockIdx.x * blockDim.x + threadIdx.x;
  const int stride = gridDim.x * blockDim.x;
  const int total = Hn * DHn * Cn;
  for (; i < total; i += stride) {
    const int h = i / (DHn * Cn);
    const int r = i - h * (DHn * Cn);
    const int d = r / Cn;
    const int c = r - d * Cn;
    out[i] = f2bf(in[((size_t)h * Cn + c) * DHn + d]);
  }
}

// ---------------- LayerNorm (fp32 in, bf16 out) ----------------

__global__ __launch_bounds__(256) void ln_kernel(const float* __restrict__ x,
                                                 const float* __restrict__ g,
                                                 const float* __restrict__ bta,
                                                 bf16_t* __restrict__ out) {
  __shared__ float red[256];
  const int row = blockIdx.x;
  const int tid = threadIdx.x;
  const float* xr = x + (size_t)row * Cn;
  float vals[4];
  float s = 0.0f;
#pragma unroll
  for (int i = 0; i < 4; ++i) {
    vals[i] = xr[tid + 256 * i];
    s += vals[i];
  }
  red[tid] = s;
  __syncthreads();
  for (int st = 128; st > 0; st >>= 1) {
    if (tid < st) red[tid] += red[tid + st];
    __syncthreads();
  }
  const float mean = red[0] * (1.0f / Cn);
  __syncthreads();
  float s2 = 0.0f;
#pragma unroll
  for (int i = 0; i < 4; ++i) {
    const float d = vals[i] - mean;
    s2 += d * d;
  }
  red[tid] = s2;
  __syncthreads();
  for (int st = 128; st > 0; st >>= 1) {
    if (tid < st) red[tid] += red[tid + st];
    __syncthreads();
  }
  const float rstd = rsqrtf(red[0] * (1.0f / Cn) + 1e-5f);
#pragma unroll
  for (int i = 0; i < 4; ++i) {
    const int c = tid + 256 * i;
    out[(size_t)row * Cn + c] = f2bf((vals[i] - mean) * rstd * g[c] + bta[c]);
  }
}

// ---------------- generic bf16 WMMA GEMM, TDM double-buffered ----------------
// C[M,N] = A[M,K] @ Bt[N,K]^T (+bias)(+ReLU)(+residual).
// Block tile 128x128; 8 waves (4x2); wave tile 32x64 -> 8 v_wmma per K step.
// Wave 0 DMA's the next A/B tiles into the spare LDS buffer via the Tensor
// Data Mover while all waves compute; one barrier per K step.

template <bool RELU, bool OUT_BF16>
__global__ __launch_bounds__(256) void gemm_bf16_kernel(
    const bf16_t* __restrict__ A, int lda, const bf16_t* __restrict__ Bt, int N,
    const float* __restrict__ bias, const float* __restrict__ residual,
    float* __restrict__ Cf, bf16_t* __restrict__ Cb, int ldc, int K) {
  constexpr int LD = 40;  // 32 + 8 pad (TDM pad fields), conflict-free + 16B aligned
  __shared__ alignas(16) bf16_t As[2][128 * LD];   // [m][k]
  __shared__ alignas(16) bf16_t Bs[2][128 * LD];   // [n][k]
  const int tid = threadIdx.x;
  const int wave = tid >> 5;
  const int wm = wave >> 1, wn = wave & 1;
  const int m0 = blockIdx.x * 128;
  const int n0 = blockIdx.y * 128;
  const int M = gridDim.x * 128;

  v8f acc[2][4];
#pragma unroll
  for (int i = 0; i < 2; ++i)
#pragma unroll
    for (int j = 0; j < 4; ++j) acc[i][j] = vzero8();

  const int nsteps = K >> 5;
  if (wave == 0) {
    tdm_load_tile((unsigned)(uintptr_t)&As[0][0], A + (size_t)m0 * lda, K, M, 32, 128, lda);
    tdm_load_tile((unsigned)(uintptr_t)&Bs[0][0], Bt + (size_t)n0 * K, K, N, 32, 128, K);
    tdm_wait0();
  }
  __syncthreads();

  for (int s = 0; s < nsteps; ++s) {
    const int cur = s & 1;
    if (s + 1 < nsteps && wave == 0) {
      const int k1 = (s + 1) << 5;
      tdm_load_tile((unsigned)(uintptr_t)&As[cur ^ 1][0], A + (size_t)m0 * lda + k1, K, M,
                    32, 128, lda);
      tdm_load_tile((unsigned)(uintptr_t)&Bs[cur ^ 1][0], Bt + (size_t)n0 * K + k1, K, N,
                    32, 128, K);
    }

    v16bf af[2], bf[4];
    af[0] = load_frag_a(&As[cur][(32 * wm) * LD], LD);
    af[1] = load_frag_a(&As[cur][(32 * wm + 16) * LD], LD);
#pragma unroll
    for (int j = 0; j < 4; ++j)
      bf[j] = load_frag_bt(&Bs[cur][(64 * wn + 16 * j) * LD], LD);
#pragma unroll
    for (int i = 0; i < 2; ++i)
#pragma unroll
      for (int j = 0; j < 4; ++j) acc[i][j] = wmma_bf16(af[i], bf[j], acc[i][j]);

    if (s + 1 < nsteps) {
      if (wave == 0) tdm_wait0();
      __syncthreads();
    }
  }

  const int lane = tid & 31;
  const int col = lane & 15, rowoff = (lane >> 4) * 8;
#pragma unroll
  for (int i = 0; i < 2; ++i)
#pragma unroll
    for (int j = 0; j < 4; ++j) {
      const int gcol = n0 + 64 * wn + 16 * j + col;
      const float bv = bias ? bias[gcol] : 0.0f;
#pragma unroll
      for (int v = 0; v < 8; ++v) {
        const int grow = m0 + 32 * wm + 16 * i + rowoff + v;
        float val = acc[i][j][v] + bv;
        if (RELU) val = fmaxf(val, 0.0f);
        if (residual) val += residual[(size_t)grow * ldc + gcol];
        if (OUT_BF16)
          Cb[(size_t)grow * ldc + gcol] = f2bf(val);
        else
          Cf[(size_t)grow * ldc + gcol] = val;
      }
    }
}

// ---------------- fused QKV projection (per head, N = DH = 64) ----------------
// Same TDM double-buffered pipeline; per-head transposed weights WT[h][d][c].

__global__ __launch_bounds__(256) void qkv_kernel(
    const bf16_t* __restrict__ Hb, const bf16_t* __restrict__ WqT,
    const bf16_t* __restrict__ WkT, const bf16_t* __restrict__ WvT,
    bf16_t* __restrict__ q, bf16_t* __restrict__ k, bf16_t* __restrict__ v) {
  constexpr int LD = 40;
  __shared__ alignas(16) bf16_t As[2][128 * LD];
  __shared__ alignas(16) bf16_t Bs[2][64 * LD];
  const int tid = threadIdx.x;
  const int wave = tid >> 5;
  const int wm = wave >> 1, wn = wave & 1;
  const int m0 = blockIdx.x * 128;
  const int h = blockIdx.y;
  const int which = blockIdx.z;
  const bf16_t* W = ((which == 0) ? WqT : (which == 1) ? WkT : WvT) + (size_t)h * DHn * Cn;
  bf16_t* out = ((which == 0) ? q : (which == 1) ? k : v) + (size_t)h * Tn * DHn;

  v8f acc[2][2];
#pragma unroll
  for (int i = 0; i < 2; ++i)
#pragma unroll
    for (int j = 0; j < 2; ++j) acc[i][j] = vzero8();

  const int nsteps = Cn >> 5;
  if (wave == 0) {
    tdm_load_tile((unsigned)(uintptr_t)&As[0][0], Hb + (size_t)m0 * Cn, Cn, BT, 32, 128, Cn);
    tdm_load_tile((unsigned)(uintptr_t)&Bs[0][0], W, Cn, DHn, 32, 64, Cn);
    tdm_wait0();
  }
  __syncthreads();

  for (int s = 0; s < nsteps; ++s) {
    const int cur = s & 1;
    if (s + 1 < nsteps && wave == 0) {
      const int k1 = (s + 1) << 5;
      tdm_load_tile((unsigned)(uintptr_t)&As[cur ^ 1][0], Hb + (size_t)m0 * Cn + k1, Cn,
                    BT, 32, 128, Cn);
      tdm_load_tile((unsigned)(uintptr_t)&Bs[cur ^ 1][0], W + k1, Cn, DHn, 32, 64, Cn);
    }

    v16bf af[2], bf[2];
    af[0] = load_frag_a(&As[cur][(32 * wm) * LD], LD);
    af[1] = load_frag_a(&As[cur][(32 * wm + 16) * LD], LD);
    bf[0] = load_frag_bt(&Bs[cur][(32 * wn) * LD], LD);
    bf[1] = load_frag_bt(&Bs[cur][(32 * wn + 16) * LD], LD);
#pragma unroll
    for (int i = 0; i < 2; ++i)
#pragma unroll
      for (int j = 0; j < 2; ++j) acc[i][j] = wmma_bf16(af[i], bf[j], acc[i][j]);

    if (s + 1 < nsteps) {
      if (wave == 0) tdm_wait0();
      __syncthreads();
    }
  }

  const int lane = tid & 31;
  const int col = lane & 15, rowoff = (lane >> 4) * 8;
#pragma unroll
  for (int i = 0; i < 2; ++i)
#pragma unroll
    for (int j = 0; j < 2; ++j) {
      const int d = 32 * wn + 16 * j + col;
#pragma unroll
      for (int v8 = 0; v8 < 8; ++v8) {
        const int grow = m0 + 32 * wm + 16 * i + rowoff + v8;  // row in [0, B*T)
        const int bb = grow >> 11;                             // /T
        const int t = grow & (Tn - 1);
        out[(size_t)bb * Hn * Tn * DHn + (size_t)t * DHn + d] = f2bf(acc[i][j][v8]);
      }
    }
}

// ---------------- flash attention (causal), bf16 WMMA ----------------
// grid (T/128, H, B); 8 waves; wave owns a 16-row q strip.
// K kept row-major [s][d] (QK^T B-frag reads contiguous d); V staged
// transposed [d][s] so the PV B-frag also reads contiguous s.

__global__ __launch_bounds__(256) void attn_kernel(const bf16_t* __restrict__ q,
                                                   const bf16_t* __restrict__ k,
                                                   const bf16_t* __restrict__ v,
                                                   bf16_t* __restrict__ att) {
  constexpr int KS_LD = 72;
  constexpr int VT_LD = 72;
  constexpr int PS_LD = 72;
  __shared__ alignas(16) bf16_t Ks[64 * KS_LD];    // [s][d]
  __shared__ alignas(16) bf16_t VsT[64 * VT_LD];   // [d][s]
  __shared__ alignas(16) bf16_t Ps[8][16 * PS_LD]; // per-wave P strip [row][s]
  const int tid = threadIdx.x;
  const int wave = tid >> 5;
  const int lane = tid & 31;
  const int col = lane & 15;
  const int hf = lane >> 4;
  const int b = blockIdx.z, h = blockIdx.y;
  const int q0 = blockIdx.x * 128;
  const size_t head_off = ((size_t)b * Hn + h) * (size_t)Tn * DHn;
  const bf16_t* qh = q + head_off;
  const bf16_t* kh = k + head_off;
  const bf16_t* vh = v + head_off;

  const int trow = q0 + wave * 16;
  v16bf aq[2];
  aq[0] = load_frag_a(qh + (size_t)trow * DHn, DHn);
  aq[1] = load_frag_a(qh + (size_t)trow * DHn + 32, DHn);

  v8f accO[4];
  float mrow[8], lrow[8];
#pragma unroll
  for (int j = 0; j < 4; ++j) accO[j] = vzero8();
#pragma unroll
  for (int v8 = 0; v8 < 8; ++v8) {
    mrow[v8] = -1e30f;
    lrow[v8] = 0.0f;
  }

  const float scale = 0.03125f;  // C^-0.5 = 1/32 (matches reference scaling)
  const int kr = tid >> 2, kc = (tid & 3) * 16;
  const int send = q0 + 128;

  for (int s0 = 0; s0 < send; s0 += 64) {
    __syncthreads();
    const uint4* gk = reinterpret_cast<const uint4*>(kh + (size_t)(s0 + kr) * DHn + kc);
    *reinterpret_cast<uint4*>(&Ks[kr * KS_LD + kc])     = gk[0];
    *reinterpret_cast<uint4*>(&Ks[kr * KS_LD + kc + 8]) = gk[1];
    const uint4* gv = reinterpret_cast<const uint4*>(vh + (size_t)(s0 + kr) * DHn + kc);
    U4B16 t0, t1;
    t0.u = gv[0];
    t1.u = gv[1];
#pragma unroll
    for (int e = 0; e < 8; ++e) {
      VsT[(kc + e) * VT_LD + kr]     = t0.e[e];
      VsT[(kc + 8 + e) * VT_LD + kr] = t1.e[e];
    }
    __syncthreads();

    // S strip = Q(16x64) @ K^T: B element (k=d, n=s) = Ks[s*KS_LD + d]
    v8f accS[4];
#pragma unroll
    for (int j = 0; j < 4; ++j) {
      accS[j] = vzero8();
      v16bf b0 = load_frag_bt(Ks + (j * 16) * KS_LD, KS_LD);       // d = 0..31
      v16bf b1 = load_frag_bt(Ks + (j * 16) * KS_LD + 32, KS_LD);  // d = 32..63
      accS[j] = wmma_bf16(aq[0], b0, accS[j]);
      accS[j] = wmma_bf16(aq[1], b1, accS[j]);
    }

    // causal mask + scale + online softmax (rows live in 16-lane half-groups)
#pragma unroll
    for (int v8 = 0; v8 < 8; ++v8) {
      const int t_idx = trow + hf * 8 + v8;
      float mx = -1e30f;
#pragma unroll
      for (int j = 0; j < 4; ++j) {
        const int s_idx = s0 + j * 16 + col;
        float sv = accS[j][v8] * scale;
        sv = (s_idx <= t_idx) ? sv : -1e30f;
        accS[j][v8] = sv;
        mx = fmaxf(mx, sv);
      }
#pragma unroll
      for (int m = 1; m <= 8; m <<= 1) mx = fmaxf(mx, __shfl_xor(mx, m, 32));
      const float mnew = fmaxf(mrow[v8], mx);
      const float corr = __expf(mrow[v8] - mnew);
      mrow[v8] = mnew;
      float rsum = 0.0f;
#pragma unroll
      for (int j = 0; j < 4; ++j) {
        const float p = __expf(accS[j][v8] - mnew);
        accS[j][v8] = p;
        rsum += p;
      }
#pragma unroll
      for (int m = 1; m <= 8; m <<= 1) rsum += __shfl_xor(rsum, m, 32);
      lrow[v8] = lrow[v8] * corr + rsum;
#pragma unroll
      for (int j2 = 0; j2 < 4; ++j2) accO[j2][v8] *= corr;
    }

    // restage P through LDS (C layout -> A layout)
#pragma unroll
    for (int j = 0; j < 4; ++j)
#pragma unroll
      for (int v8 = 0; v8 < 8; ++v8)
        Ps[wave][(hf * 8 + v8) * PS_LD + j * 16 + col] = f2bf(accS[j][v8]);
    __syncthreads();

    v16bf ap0 = load_frag_a(Ps[wave], PS_LD);
    v16bf ap1 = load_frag_a(Ps[wave] + 32, PS_LD);
    // O += P @ V: B element (k=s, n=d) = VsT[d*VT_LD + s]
#pragma unroll
    for (int j2 = 0; j2 < 4; ++j2) {
      v16bf bv0 = load_frag_bt(VsT + (j2 * 16) * VT_LD, VT_LD);       // s = 0..31
      v16bf bv1 = load_frag_bt(VsT + (j2 * 16) * VT_LD + 32, VT_LD);  // s = 32..63
      accO[j2] = wmma_bf16(ap0, bv0, accO[j2]);
      accO[j2] = wmma_bf16(ap1, bv1, accO[j2]);
    }
  }

  // finalize: O / l, write heads-concatenated [B,T,C] bf16
#pragma unroll
  for (int j2 = 0; j2 < 4; ++j2)
#pragma unroll
    for (int v8 = 0; v8 < 8; ++v8) {
      const int t_idx = trow + hf * 8 + v8;
      const int d = j2 * 16 + col;
      att[((size_t)b * Tn + t_idx) * Cn + h * DHn + d] = f2bf(accO[j2][v8] / lrow[v8]);
    }
}

// ---------------- host launcher ----------------

extern "C" void kernel_launch(void* const* d_in, const int* in_sizes, int n_in,
                              void* d_out, int out_size, void* d_ws, size_t ws_size,
                              hipStream_t stream) {
  (void)in_sizes; (void)n_in; (void)out_size; (void)ws_size;
  const float* x    = (const float*)d_in[0];
  const float* ln1g = (const float*)d_in[1];
  const float* ln1b = (const float*)d_in[2];
  const float* Wq   = (const float*)d_in[3];
  const float* Wk   = (const float*)d_in[4];
  const float* Wv   = (const float*)d_in[5];
  const float* Wo   = (const float*)d_in[6];
  const float* bo   = (const float*)d_in[7];
  const float* ln2g = (const float*)d_in[8];
  const float* ln2b = (const float*)d_in[9];
  const float* W1   = (const float*)d_in[10];
  const float* b1   = (const float*)d_in[11];
  const float* W2   = (const float*)d_in[12];
  const float* b2   = (const float*)d_in[13];
  float* out = (float*)d_out;

  char* ws = (char*)d_ws;
  size_t off = 0;
  auto alloc = [&](size_t bytes) -> void* {
    void* p = ws + off;
    off += (bytes + 255) & ~(size_t)255;
    return p;
  };

  // transposed bf16 weights (n-major) for TDM-friendly B tiles
  bf16_t* wqt = (bf16_t*)alloc((size_t)Hn * DHn * Cn * 2);
  bf16_t* wkt = (bf16_t*)alloc((size_t)Hn * DHn * Cn * 2);
  bf16_t* wvt = (bf16_t*)alloc((size_t)Hn * DHn * Cn * 2);
  bf16_t* wot = (bf16_t*)alloc((size_t)Cn * Cn * 2);
  bf16_t* w1t = (bf16_t*)alloc((size_t)HIDn * Cn * 2);
  bf16_t* w2t = (bf16_t*)alloc((size_t)Cn * HIDn * 2);
  bf16_t* hb  = (bf16_t*)alloc((size_t)BT * Cn * 2);
  bf16_t* qb  = (bf16_t*)alloc((size_t)Bsz * Hn * Tn * DHn * 2);
  bf16_t* kb  = (bf16_t*)alloc((size_t)Bsz * Hn * Tn * DHn * 2);
  bf16_t* vb  = (bf16_t*)alloc((size_t)Bsz * Hn * Tn * DHn * 2);
  bf16_t* attb = (bf16_t*)alloc((size_t)BT * Cn * 2);
  float*  x2  = (float*)alloc((size_t)BT * Cn * 4);
  bf16_t* h2  = (bf16_t*)alloc((size_t)BT * Cn * 2);
  bf16_t* a1  = (bf16_t*)alloc((size_t)BT * HIDn * 2);

  // 1) weights -> bf16, transposed to n-major
  tr_head_bf16_kernel<<<512, 256, 0, stream>>>(Wq, wqt);
  tr_head_bf16_kernel<<<512, 256, 0, stream>>>(Wk, wkt);
  tr_head_bf16_kernel<<<512, 256, 0, stream>>>(Wv, wvt);
  tr2d_bf16_kernel<<<512, 256, 0, stream>>>(Wo, wot, Cn, Cn);
  tr2d_bf16_kernel<<<512, 256, 0, stream>>>(W1, w1t, Cn, HIDn);
  tr2d_bf16_kernel<<<512, 256, 0, stream>>>(W2, w2t, HIDn, Cn);

  // 2) ln1
  ln_kernel<<<BT, 256, 0, stream>>>(x, ln1g, ln1b, hb);

  // 3) q,k,v = h @ {Wq,Wk,Wv}   -> [B,H,T,DH] bf16
  qkv_kernel<<<dim3(BT / 128, Hn, 3), 256, 0, stream>>>(hb, wqt, wkt, wvt, qb, kb, vb);

  // 4) causal flash attention    -> att [B,T,C] bf16 (heads concatenated)
  attn_kernel<<<dim3(Tn / 128, Hn, Bsz), 256, 0, stream>>>(qb, kb, vb, attb);

  // 5) x2 = x + att @ Wo + bo    -> fp32
  gemm_bf16_kernel<false, false><<<dim3(BT / 128, Cn / 128), 256, 0, stream>>>(
      attb, Cn, wot, Cn, bo, x, x2, nullptr, Cn, Cn);

  // 6) ln2
  ln_kernel<<<BT, 256, 0, stream>>>(x2, ln2g, ln2b, h2);

  // 7) a1 = relu(h2 @ W1 + b1)   -> bf16
  gemm_bf16_kernel<true, true><<<dim3(BT / 128, HIDn / 128), 256, 0, stream>>>(
      h2, Cn, w1t, HIDn, b1, nullptr, nullptr, a1, HIDn, Cn);

  // 8) out = x2 + a1 @ W2 + b2   -> fp32
  gemm_bf16_kernel<false, false><<<dim3(BT / 128, Cn / 128), 256, 0, stream>>>(
      a1, HIDn, w2t, Cn, b2, x2, out, nullptr, Cn, HIDn);
}
